// GNN_F_28527172780190
// MI455X (gfx1250) — matrix operand order repaired and checked by reading, not compiled, not run on hardware
//
#include <hip/hip_runtime.h>
#include <hip/hip_bf16.h>

typedef __attribute__((ext_vector_type(16))) _Float16 v16h;
typedef __attribute__((ext_vector_type(8)))  _Float16 v8h;
typedef __attribute__((ext_vector_type(8)))  float    v8f;

#define N_NODES 50000
#define N_EDGES 1600000

// ---------------------------------------------------------------------------
// small utility kernels
// ---------------------------------------------------------------------------
__global__ void fill_kernel(float* __restrict__ p, float v, int n) {
    int i = blockIdx.x * blockDim.x + threadIdx.x;
    if (i < n) p[i] = v;
}

__global__ void deg_kernel(const int* __restrict__ dst, float* __restrict__ deg, int nE) {
    int i = blockIdx.x * blockDim.x + threadIdx.x;
    if (i < nE) atomicAdd(&deg[dst[i]], 1.0f);
}

__global__ void rsqrt_kernel(float* __restrict__ p, int n) {
    int i = blockIdx.x * blockDim.x + threadIdx.x;
    if (i < n) p[i] = rsqrtf(p[i]);
}

__global__ void cast_f16_kernel(const float* __restrict__ in, _Float16* __restrict__ out,
                                long long n) {
    long long i = (long long)blockIdx.x * blockDim.x + threadIdx.x;
    if (i < n) out[i] = (_Float16)in[i];
}

// W [K x M] f32 row-major  ->  Wt [M x K] f16 row-major
__global__ void transpose_cast_kernel(const float* __restrict__ W, _Float16* __restrict__ Wt,
                                      int K, int M) {
    int m = blockIdx.x * 16 + threadIdx.x;
    int k = blockIdx.y * 16 + threadIdx.y;
    if (k < K && m < M) Wt[(size_t)m * K + k] = (_Float16)W[(size_t)k * M + m];
}

// ---------------------------------------------------------------------------
// WMMA GEMM:  C[N x M] (f32) = A[N x K] (f16) * Bt[M x K]^T (f16)  (+ bias)
// block = 256 threads = 8 waves; wave w -> rows (bx*8+w)*16 .. +15, cols by*64 .. +63
// ---------------------------------------------------------------------------
__global__ __launch_bounds__(256)
void wmma_gemm_kernel(const _Float16* __restrict__ A,
                      const _Float16* __restrict__ Bt,
                      const float* __restrict__ bias,   // may be nullptr
                      float* __restrict__ C,
                      int N, int K, int M) {
    const int lane = threadIdx.x & 31;
    const int wave = threadIdx.x >> 5;
    const int lr   = lane & 15;
    const int half = lane >> 4;
    const int rowBase = (blockIdx.x * 8 + wave) * 16;
    const int colBase = blockIdx.y * 64;
    if (rowBase >= N) return;

    v8f acc[4] = {v8f{}, v8f{}, v8f{}, v8f{}};

    const _Float16* arow = A + (size_t)(rowBase + lr) * K;
    const _Float16* brow[4];
#pragma unroll
    for (int t = 0; t < 4; ++t)
        brow[t] = Bt + (size_t)(colBase + t * 16 + lr) * K;

    for (int k0 = 0; k0 < K; k0 += 32) {
        // A fragment: contiguous layout — lane half selects K+0..15 / K+16..31
        // (two contiguous 16B loads).
        v8h alo = *(const v8h*)(arow + k0 + half * 16);
        v8h ahi = *(const v8h*)(arow + k0 + half * 16 + 8);
        v16h a;
#pragma unroll
        for (int i = 0; i < 8; ++i) { a[i] = alo[i]; a[8 + i] = ahi[i]; }

        const int kb = k0 + half * 16;   // B blocked layout: one 32B load
#pragma unroll
        for (int t = 0; t < 4; ++t) {
            v16h b = *(const v16h*)(brow[t] + kb);
            acc[t] = __builtin_amdgcn_wmma_f32_16x16x32_f16(
                false, a, false, b, (short)0, acc[t], false, false);
        }
    }

#pragma unroll
    for (int t = 0; t < 4; ++t) {
        const int col = colBase + t * 16 + lr;
        const float bv = bias ? bias[col] : 0.0f;
#pragma unroll
        for (int r = 0; r < 8; ++r) {
            const int row = rowBase + r + half * 8;   // C layout: lanes16-31 -> M=r+8
            C[(size_t)row * M + col] = acc[t][r] + bv;
        }
    }
}

// ---------------------------------------------------------------------------
// edge scatter: acc[dst] += dis[src]*dis[dst] * g[src]   (one wave per edge)
// acc already holds the linear-branch output, so no zero-init pass needed.
// ---------------------------------------------------------------------------
__global__ __launch_bounds__(256)
void scatter_kernel(const int* __restrict__ src, const int* __restrict__ dst,
                    const float* __restrict__ dis, const float* __restrict__ g,
                    float* __restrict__ acc, int D, int nE) {
    int gtid = blockIdx.x * blockDim.x + threadIdx.x;
    int wid  = gtid >> 5;
    int lane = gtid & 31;
    int nw   = (gridDim.x * blockDim.x) >> 5;
    for (int e = wid; e < nE; e += nw) {
        int s = src[e], d = dst[e];
        float norm = dis[s] * dis[d];
        const float* gs = g + (size_t)s * D;
        float* ad = acc + (size_t)d * D;
        for (int j = lane; j < D; j += 32)
            atomicAdd(&ad[j], norm * gs[j]);
    }
}

// ---------------------------------------------------------------------------
// combine + ReLU + instance-norm (ddof=1) + f16 recast.  One wave per row, D=512.
// linagg = lin_out + segment_sum already; add self-loop dis^2*g + b_gcn.
// hOut (f32, may be null -> only needed for penultimate), h16Out always written.
// ---------------------------------------------------------------------------
__global__ __launch_bounds__(256)
void combine_norm_kernel(const float* __restrict__ linagg, const float* __restrict__ g,
                         const float* __restrict__ dis, const float* __restrict__ bg,
                         float* __restrict__ hOut, _Float16* __restrict__ h16Out, int N) {
    const int D = 512;
    int lane = threadIdx.x & 31;
    int row  = (blockIdx.x * blockDim.x + threadIdx.x) >> 5;
    if (row >= N) return;
    float sw = dis[row]; sw *= sw;
    const float* la = linagg + (size_t)row * D;
    const float* gr = g      + (size_t)row * D;

    float v[16];
    float sum = 0.0f;
#pragma unroll
    for (int t = 0; t < 16; ++t) {
        int j = lane + t * 32;
        float x = la[j] + sw * gr[j] + bg[j];
        x = fmaxf(x, 0.0f);              // ReLU
        v[t] = x; sum += x;
    }
#pragma unroll
    for (int off = 16; off > 0; off >>= 1) sum += __shfl_xor(sum, off, 32);
    float mean = sum * (1.0f / 512.0f);
    float sq = 0.0f;
#pragma unroll
    for (int t = 0; t < 16; ++t) { float d = v[t] - mean; sq += d * d; }
#pragma unroll
    for (int off = 16; off > 0; off >>= 1) sq += __shfl_xor(sq, off, 32);
    float inv = 1.0f / (sqrtf(sq * (1.0f / 511.0f)) + 1e-5f);   // unbiased std

    float*    ho  = hOut ? hOut + (size_t)row * D : nullptr;
    _Float16* h16 = h16Out + (size_t)row * D;
#pragma unroll
    for (int t = 0; t < 16; ++t) {
        int j = lane + t * 32;
        float o = (v[t] - mean) * inv;
        if (ho) ho[j] = o;
        h16[j] = (_Float16)o;
    }
}

// final layer: out = linagg + dis^2*g + b_gcn  (no relu/norm), D=256
__global__ void final_combine_kernel(const float* __restrict__ linagg,
                                     const float* __restrict__ g,
                                     const float* __restrict__ dis,
                                     const float* __restrict__ bg,
                                     float* __restrict__ out, int N, int D) {
    long long i = (long long)blockIdx.x * blockDim.x + threadIdx.x;
    long long total = (long long)N * D;
    if (i >= total) return;
    int row = (int)(i / D), col = (int)(i % D);
    float sw = dis[row]; sw *= sw;
    out[i] = linagg[i] + sw * g[i] + bg[col];
}

// ---------------------------------------------------------------------------
// launch
// ---------------------------------------------------------------------------
extern "C" void kernel_launch(void* const* d_in, const int* in_sizes, int n_in,
                              void* d_out, int out_size, void* d_ws, size_t ws_size,
                              hipStream_t stream) {
    const int N = N_NODES, E = N_EDGES;
    const float* x      = (const float*)d_in[0];
    const int*   ei     = (const int*)  d_in[1];
    const float* W1_lin = (const float*)d_in[2];
    const float* b1_lin = (const float*)d_in[3];
    const float* W1_gcn = (const float*)d_in[4];
    const float* b1_gcn = (const float*)d_in[5];
    const float* Wm_lin = (const float*)d_in[6];
    const float* bm_lin = (const float*)d_in[7];
    const float* Wm_gcn = (const float*)d_in[8];
    const float* bm_gcn = (const float*)d_in[9];
    const float* Wo_lin = (const float*)d_in[10];
    const float* bo_lin = (const float*)d_in[11];
    const float* Wo_gcn = (const float*)d_in[12];
    const float* bo_gcn = (const float*)d_in[13];
    const int* src = ei;
    const int* dst = ei + E;

    // workspace carve-out
    char* w = (char*)d_ws;
    auto carve = [&](size_t bytes) { char* p = w; w += (bytes + 255) & ~(size_t)255; return p; };
    float*    dis  = (float*)   carve((size_t)N * 4);
    _Float16* x16  = (_Float16*)carve((size_t)N * 256 * 2);
    _Float16* h16  = (_Float16*)carve((size_t)N * 512 * 2);
    float*    g    = (float*)   carve((size_t)N * 512 * 4);
    float*    l    = (float*)   carve((size_t)N * 512 * 4);
    _Float16* w1l  = (_Float16*)carve((size_t)512 * 256 * 2);
    _Float16* w1g  = (_Float16*)carve((size_t)512 * 256 * 2);
    _Float16* wml  = (_Float16*)carve((size_t)2 * 512 * 512 * 2);
    _Float16* wmg  = (_Float16*)carve((size_t)2 * 512 * 512 * 2);
    _Float16* wol  = (_Float16*)carve((size_t)256 * 512 * 2);
    _Float16* wog  = (_Float16*)carve((size_t)256 * 512 * 2);

    float* out_final = (float*)d_out;                       // [N x 256]
    float* penult    = (float*)d_out + (size_t)N * 256;     // [N x 512]

    // --- degree / dis ---
    fill_kernel<<<(N + 255) / 256, 256, 0, stream>>>(dis, 1.0f, N);     // self-loop
    deg_kernel<<<(E + 255) / 256, 256, 0, stream>>>(dst, dis, E);
    rsqrt_kernel<<<(N + 255) / 256, 256, 0, stream>>>(dis, N);

    // --- weights -> f16 transposed [M x K] ---
    dim3 tb(16, 16);
    transpose_cast_kernel<<<dim3(512 / 16, 256 / 16), tb, 0, stream>>>(W1_lin, w1l, 256, 512);
    transpose_cast_kernel<<<dim3(512 / 16, 256 / 16), tb, 0, stream>>>(W1_gcn, w1g, 256, 512);
    for (int i = 0; i < 2; ++i) {
        transpose_cast_kernel<<<dim3(32, 32), tb, 0, stream>>>(
            Wm_lin + (size_t)i * 512 * 512, wml + (size_t)i * 512 * 512, 512, 512);
        transpose_cast_kernel<<<dim3(32, 32), tb, 0, stream>>>(
            Wm_gcn + (size_t)i * 512 * 512, wmg + (size_t)i * 512 * 512, 512, 512);
    }
    transpose_cast_kernel<<<dim3(256 / 16, 512 / 16), tb, 0, stream>>>(Wo_lin, wol, 512, 256);
    transpose_cast_kernel<<<dim3(256 / 16, 512 / 16), tb, 0, stream>>>(Wo_gcn, wog, 512, 256);

    // --- x -> f16 ---
    {
        long long n = (long long)N * 256;
        cast_f16_kernel<<<(unsigned)((n + 255) / 256), 256, 0, stream>>>(x, x16, n);
    }

    const int rowTiles  = N / 16;                       // 3125
    const int gemmBlksX = (rowTiles + 7) / 8;           // 391
    dim3 gemmBlk(256);
    const int scatterGrid = 4096;
    const int normGrid = (N + 7) / 8;                   // 8 rows (waves) per block

    // --- layer 1 (K=256 -> M=512) ---
    wmma_gemm_kernel<<<dim3(gemmBlksX, 512 / 64), gemmBlk, 0, stream>>>(x16, w1l, b1_lin, l, N, 256, 512);
    wmma_gemm_kernel<<<dim3(gemmBlksX, 512 / 64), gemmBlk, 0, stream>>>(x16, w1g, nullptr, g, N, 256, 512);
    scatter_kernel<<<scatterGrid, 256, 0, stream>>>(src, dst, dis, g, l, 512, E);
    combine_norm_kernel<<<normGrid, 256, 0, stream>>>(l, g, dis, b1_gcn, nullptr, h16, N);

    // --- mid layers (K=512 -> M=512) ---
    for (int i = 0; i < 2; ++i) {
        const _Float16* Wl = wml + (size_t)i * 512 * 512;
        const _Float16* Wg = wmg + (size_t)i * 512 * 512;
        wmma_gemm_kernel<<<dim3(gemmBlksX, 512 / 64), gemmBlk, 0, stream>>>(h16, Wl, bm_lin + (size_t)i * 512, l, N, 512, 512);
        wmma_gemm_kernel<<<dim3(gemmBlksX, 512 / 64), gemmBlk, 0, stream>>>(h16, Wg, nullptr, g, N, 512, 512);
        scatter_kernel<<<scatterGrid, 256, 0, stream>>>(src, dst, dis, g, l, 512, E);
        // last mid layer writes penultimate f32 into d_out
        combine_norm_kernel<<<normGrid, 256, 0, stream>>>(
            l, g, dis, bm_gcn + (size_t)i * 512, (i == 1) ? penult : nullptr, h16, N);
    }

    // --- output layer (K=512 -> M=256) ---
    wmma_gemm_kernel<<<dim3(gemmBlksX, 256 / 64), gemmBlk, 0, stream>>>(h16, wol, bo_lin, l, N, 512, 256);
    wmma_gemm_kernel<<<dim3(gemmBlksX, 256 / 64), gemmBlk, 0, stream>>>(h16, wog, nullptr, g, N, 512, 256);
    scatter_kernel<<<scatterGrid, 256, 0, stream>>>(src, dst, dis, g, l, 256, E);
    {
        long long n = (long long)N * 256;
        final_combine_kernel<<<(unsigned)((n + 255) / 256), 256, 0, stream>>>(l, g, dis, bo_gcn, out_final, N, 256);
    }
}
